// ConvGRU_13142599926373
// MI455X (gfx1250) — matrix operand order repaired and checked by reading, not compiled, not run on hardware
//
#include <hip/hip_runtime.h>
#include <cstdint>

#define N_PTS  120000
#define HID    128
#define INP    320
#define CIN    448
#define KOFFS  27
#define KTOT   28          // 27 conv offsets + 1 point-transform
#define NCH    14          // CIN / 32 (bf16 WMMA K=32)
#define CTILES 8           // HID / 16
#define TILE_M 128
#define LDS_STRIDE 40      // halfs per row (80 bytes, 16B aligned, bank spread)

typedef __attribute__((ext_vector_type(16))) __bf16 v16bf;
typedef __attribute__((ext_vector_type(8)))  float  v8f;

union FragU { uint4 u[2]; v16bf v; };

static __device__ __forceinline__ unsigned short f2bf(float f) {
  unsigned int u = __float_as_uint(f);
  unsigned int r = u + 0x7FFFu + ((u >> 16) & 1u);   // round-to-nearest-even
  return (unsigned short)(r >> 16);
}

// Async gather: 16 bytes global -> LDS, tracked by ASYNCcnt (CDNA5 §15.18.3 op 98).
static __device__ __forceinline__ void async_ld16(unsigned lds_off,
                                                  unsigned long long gaddr) {
  asm volatile("global_load_async_to_lds_b128 %0, %1, off"
               :: "v"(lds_off), "v"(gaddr) : "memory");
}
static __device__ __forceinline__ void wait_async0() {
  asm volatile("s_wait_asynccnt 0x0" ::: "memory");
}

// Base byte-address of the gathered feature row for (row, k); zero page if masked.
static __device__ __forceinline__ unsigned long long gather_base(
    const unsigned short* __restrict__ feat,
    const unsigned short* __restrict__ zpage,
    const int* __restrict__ nbr, int row, int k) {
  int idx = -1;
  if (row < N_PTS) idx = (k < KOFFS) ? nbr[(size_t)row * KOFFS + k] : row;
  const unsigned short* p = (idx >= 0) ? (feat + (size_t)idx * CIN) : zpage;
  return (unsigned long long)(uintptr_t)p;
}

// ---------------------------------------------------------------- stage 1
__global__ void convert_features(const float* __restrict__ h,
                                 const float* __restrict__ x,
                                 unsigned short* __restrict__ hxbf,
                                 unsigned short* __restrict__ rxbf,
                                 unsigned short* __restrict__ zpage) {
  size_t i = (size_t)blockIdx.x * blockDim.x + threadIdx.x;
  if (i >= (size_t)N_PTS * CIN) return;
  if (i < 512) zpage[i] = 0;      // 1KB zero page for masked gathers
  int c = (int)(i % CIN);
  size_t n = i / CIN;
  float v = (c < HID) ? h[n * HID + c] : x[n * INP + (c - HID)];
  unsigned short us = f2bf(v);
  hxbf[i] = us;
  if (c >= HID) rxbf[i] = us;     // x-half of rx is static
}

// ---------------------------------------------------------------- stage 2
// Pack W[k] (CINxHID, k=27 -> point transform) into per-lane WMMA-B fragments:
// frag f = (k*NCH + c)*CTILES + t ; lane l holds 16 contiguous bf16:
//   element j = W[c*32 + (l<16?0:8) + j + (j&8)][t*16 + (l&15)]
__global__ void pack_weights(const float* __restrict__ Wz, const float* __restrict__ Wzp,
                             const float* __restrict__ Wr, const float* __restrict__ Wrp,
                             const float* __restrict__ Wq, const float* __restrict__ Wqp,
                             unsigned short* __restrict__ pz,
                             unsigned short* __restrict__ pr,
                             unsigned short* __restrict__ pq) {
  const int perGate = KTOT * NCH * CTILES * 32;      // 100352
  int i = blockIdx.x * blockDim.x + threadIdx.x;
  if (i >= 3 * perGate) return;
  int gate = i / perGate;
  int rem  = i % perGate;
  int lane = rem & 31;
  int f    = rem >> 5;
  int t    = f & 7;
  int c    = (f >> 3) % NCH;
  int k    = f / (NCH * CTILES);
  const float* W  = (gate == 0) ? Wz  : (gate == 1) ? Wr  : Wq;
  const float* Wp = (gate == 0) ? Wzp : (gate == 1) ? Wrp : Wqp;
  unsigned short* dst =
      ((gate == 0) ? pz : (gate == 1) ? pr : pq) + (size_t)f * 512 + lane * 16;
  int koff = (lane < 16) ? 0 : 8;
  int col  = t * 16 + (lane & 15);
  #pragma unroll
  for (int j = 0; j < 16; ++j) {
    int kk = c * 32 + koff + j + (j & 8);
    float v = (k < KOFFS) ? W[((size_t)k * CIN + kk) * HID + col]
                          : Wp[(size_t)kk * HID + col];
    dst[j] = f2bf(v);
  }
}

// ---------------------------------------------------------------- stage 3
// z,r gates: async double-buffered gathers, 2 WMMAs per A fragment.
// Stash z in d_out, write r*h half of rx (bf16).
__global__ __launch_bounds__(256)
void gates_zr(const unsigned short* __restrict__ hxbf,
              const int* __restrict__ nbr,
              const unsigned short* __restrict__ pz,
              const unsigned short* __restrict__ pr,
              const float* __restrict__ bz,
              const float* __restrict__ br,
              const float* __restrict__ h,
              const unsigned short* __restrict__ zpage,
              float* __restrict__ zout,
              unsigned short* __restrict__ rxbf) {
  __shared__ alignas(16) unsigned short ldsA[2][TILE_M * LDS_STRIDE];
  const int tid  = threadIdx.x;
  const int lane = tid & 31;
  const int wave = tid >> 5;
  const int rowBase = blockIdx.x * TILE_M;

  v8f accZ[8]; v8f accR[8];
  #pragma unroll
  for (int i = 0; i < 8; ++i) { accZ[i] = (v8f){}; accR[i] = (v8f){}; }

  // gather assignment: thread -> rows (tid>>2) and (tid>>2)+64, 16B slice part
  const int gr   = tid >> 2;
  const int part = tid & 3;
  const int row0 = rowBase + gr;
  const int row1 = row0 + 64;
  const unsigned pb = (unsigned)part * 16;   // byte offset within 64B row chunk
  unsigned l0[2], l1[2];
  l0[0] = (unsigned)(uintptr_t)&ldsA[0][gr * LDS_STRIDE + part * 8];
  l0[1] = (unsigned)(uintptr_t)&ldsA[1][gr * LDS_STRIDE + part * 8];
  l1[0] = (unsigned)(uintptr_t)&ldsA[0][(gr + 64) * LDS_STRIDE + part * 8];
  l1[1] = (unsigned)(uintptr_t)&ldsA[1][(gr + 64) * LDS_STRIDE + part * 8];

  const int aoffLo = (lane < 16) ? 0 : 8;    // halfs
  const int r0 = lane & 15;

  unsigned long long g0 = gather_base(hxbf, zpage, nbr, row0, 0);
  unsigned long long g1 = gather_base(hxbf, zpage, nbr, row1, 0);
  async_ld16(l0[0], g0 + pb);                // prologue: chunk 0 -> buf 0
  async_ld16(l1[0], g1 + pb);

  for (int k = 0; k < KTOT; ++k) {
    for (int c = 0; c < NCH; ++c) {
      const int buf = c & 1;
      wait_async0();            // own async writes for chunk c landed
      __syncthreads();          // all waves: chunk c visible, buf^1 reads done
      if (c + 1 < NCH) {
        async_ld16(l0[buf ^ 1], g0 + (unsigned)(c + 1) * 64 + pb);
        async_ld16(l1[buf ^ 1], g1 + (unsigned)(c + 1) * 64 + pb);
      } else if (k + 1 < KTOT) {
        g0 = gather_base(hxbf, zpage, nbr, row0, k + 1);
        g1 = gather_base(hxbf, zpage, nbr, row1, k + 1);
        async_ld16(l0[buf ^ 1], g0 + pb);
        async_ld16(l1[buf ^ 1], g1 + pb);
      }
      FragU fbz, fbr;
      {
        size_t f = ((size_t)k * NCH + c) * CTILES + wave;
        const uint4* gz = (const uint4*)(pz + f * 512 + lane * 16);
        const uint4* gq = (const uint4*)(pr + f * 512 + lane * 16);
        fbz.u[0] = gz[0]; fbz.u[1] = gz[1];
        fbr.u[0] = gq[0]; fbr.u[1] = gq[1];
      }
      #pragma unroll
      for (int rt = 0; rt < 8; ++rt) {
        FragU a;
        const unsigned short* base = &ldsA[buf][(rt * 16 + r0) * LDS_STRIDE];
        a.u[0] = *(const uint4*)(base + aoffLo);        // K lo 8
        a.u[1] = *(const uint4*)(base + aoffLo + 16);   // K hi 8
        accZ[rt] = __builtin_amdgcn_wmma_f32_16x16x32_bf16(
            false, a.v, false, fbz.v, (short)0, accZ[rt], false, false);
        accR[rt] = __builtin_amdgcn_wmma_f32_16x16x32_bf16(
            false, a.v, false, fbr.v, (short)0, accR[rt], false, false);
      }
    }
  }
  // epilogue: sigmoid, z -> d_out (temp), r*h -> rx bf16 (row-local)
  int col = wave * 16 + (lane & 15);
  float bzv = bz[col], brv = br[col];
  int eoff = (lane >> 4) << 3;
  for (int rt = 0; rt < 8; ++rt) {
    #pragma unroll
    for (int e = 0; e < 8; ++e) {
      int row = rowBase + rt * 16 + e + eoff;
      if (row < N_PTS) {
        float zv = 1.f / (1.f + __expf(-(accZ[rt][e] + bzv)));
        float rv = 1.f / (1.f + __expf(-(accR[rt][e] + brv)));
        zout[(size_t)row * HID + col] = zv;
        float hv = h[(size_t)row * HID + col];
        rxbf[(size_t)row * CIN + col] = f2bf(rv * hv);
      }
    }
  }
}

// ---------------------------------------------------------------- stage 4
__global__ __launch_bounds__(256)
void gate_q_combine(const unsigned short* __restrict__ rxbf,
                    const int* __restrict__ nbr,
                    const unsigned short* __restrict__ pq,
                    const float* __restrict__ bq,
                    const float* __restrict__ h,
                    const unsigned short* __restrict__ zpage,
                    float* __restrict__ out) {   // holds z on entry
  __shared__ alignas(16) unsigned short ldsA[2][TILE_M * LDS_STRIDE];
  const int tid  = threadIdx.x;
  const int lane = tid & 31;
  const int wave = tid >> 5;
  const int rowBase = blockIdx.x * TILE_M;

  v8f accQ[8];
  #pragma unroll
  for (int i = 0; i < 8; ++i) accQ[i] = (v8f){};

  const int gr   = tid >> 2;
  const int part = tid & 3;
  const int row0 = rowBase + gr;
  const int row1 = row0 + 64;
  const unsigned pb = (unsigned)part * 16;
  unsigned l0[2], l1[2];
  l0[0] = (unsigned)(uintptr_t)&ldsA[0][gr * LDS_STRIDE + part * 8];
  l0[1] = (unsigned)(uintptr_t)&ldsA[1][gr * LDS_STRIDE + part * 8];
  l1[0] = (unsigned)(uintptr_t)&ldsA[0][(gr + 64) * LDS_STRIDE + part * 8];
  l1[1] = (unsigned)(uintptr_t)&ldsA[1][(gr + 64) * LDS_STRIDE + part * 8];

  const int aoffLo = (lane < 16) ? 0 : 8;
  const int r0 = lane & 15;

  unsigned long long g0 = gather_base(rxbf, zpage, nbr, row0, 0);
  unsigned long long g1 = gather_base(rxbf, zpage, nbr, row1, 0);
  async_ld16(l0[0], g0 + pb);
  async_ld16(l1[0], g1 + pb);

  for (int k = 0; k < KTOT; ++k) {
    for (int c = 0; c < NCH; ++c) {
      const int buf = c & 1;
      wait_async0();
      __syncthreads();
      if (c + 1 < NCH) {
        async_ld16(l0[buf ^ 1], g0 + (unsigned)(c + 1) * 64 + pb);
        async_ld16(l1[buf ^ 1], g1 + (unsigned)(c + 1) * 64 + pb);
      } else if (k + 1 < KTOT) {
        g0 = gather_base(rxbf, zpage, nbr, row0, k + 1);
        g1 = gather_base(rxbf, zpage, nbr, row1, k + 1);
        async_ld16(l0[buf ^ 1], g0 + pb);
        async_ld16(l1[buf ^ 1], g1 + pb);
      }
      FragU fbq;
      {
        size_t f = ((size_t)k * NCH + c) * CTILES + wave;
        const uint4* gq = (const uint4*)(pq + f * 512 + lane * 16);
        fbq.u[0] = gq[0]; fbq.u[1] = gq[1];
      }
      #pragma unroll
      for (int rt = 0; rt < 8; ++rt) {
        FragU a;
        const unsigned short* base = &ldsA[buf][(rt * 16 + r0) * LDS_STRIDE];
        a.u[0] = *(const uint4*)(base + aoffLo);
        a.u[1] = *(const uint4*)(base + aoffLo + 16);
        accQ[rt] = __builtin_amdgcn_wmma_f32_16x16x32_bf16(
            false, a.v, false, fbq.v, (short)0, accQ[rt], false, false);
      }
    }
  }
  // epilogue: q = tanh(.); out = (1-z)*h + z*q
  int col = wave * 16 + (lane & 15);
  float bqv = bq[col];
  int eoff = (lane >> 4) << 3;
  for (int rt = 0; rt < 8; ++rt) {
    #pragma unroll
    for (int e = 0; e < 8; ++e) {
      int row = rowBase + rt * 16 + e + eoff;
      if (row < N_PTS) {
        size_t o = (size_t)row * HID + col;
        float qv = tanhf(accQ[rt][e] + bqv);
        float zv = out[o];
        float hv = h[o];
        out[o] = (1.f - zv) * hv + zv * qv;
      }
    }
  }
}

// ---------------------------------------------------------------- launch
extern "C" void kernel_launch(void* const* d_in, const int* in_sizes, int n_in,
                              void* d_out, int out_size, void* d_ws, size_t ws_size,
                              hipStream_t stream) {
  const float* h   = (const float*)d_in[0];
  const float* x   = (const float*)d_in[1];
  const int*   nbr = (const int*)  d_in[2];
  const float* Wz  = (const float*)d_in[3];
  const float* Wzp = (const float*)d_in[4];
  const float* bz  = (const float*)d_in[5];
  const float* Wr  = (const float*)d_in[6];
  const float* Wrp = (const float*)d_in[7];
  const float* br  = (const float*)d_in[8];
  const float* Wq  = (const float*)d_in[9];
  const float* Wqp = (const float*)d_in[10];
  const float* bq  = (const float*)d_in[11];
  float* out = (float*)d_out;

  char* ws = (char*)d_ws;
  const size_t featBytes = (size_t)N_PTS * CIN * 2;                 // 107.52 MB
  const size_t packBytes = (size_t)KTOT * NCH * CTILES * 512 * 2;   // 3.21 MB
  unsigned short* hxbf = (unsigned short*)ws;
  unsigned short* rxbf = (unsigned short*)(ws + featBytes);
  unsigned short* pz   = (unsigned short*)(ws + 2 * featBytes);
  unsigned short* pr   = (unsigned short*)(ws + 2 * featBytes + packBytes);
  unsigned short* pq   = (unsigned short*)(ws + 2 * featBytes + 2 * packBytes);
  unsigned short* zpg  = (unsigned short*)(ws + 2 * featBytes + 3 * packBytes);

  {
    size_t T = (size_t)N_PTS * CIN;
    convert_features<<<(unsigned)((T + 255) / 256), 256, 0, stream>>>(h, x, hxbf,
                                                                      rxbf, zpg);
  }
  {
    int T = 3 * KTOT * NCH * CTILES * 32;
    pack_weights<<<(T + 255) / 256, 256, 0, stream>>>(Wz, Wzp, Wr, Wrp, Wq, Wqp,
                                                      pz, pr, pq);
  }
  int blocks = (N_PTS + TILE_M - 1) / TILE_M;   // 938
  gates_zr<<<blocks, 256, 0, stream>>>(hxbf, nbr, pz, pr, bz, br, h, zpg, out, rxbf);
  gate_q_combine<<<blocks, 256, 0, stream>>>(rxbf, nbr, pq, bq, h, zpg, out);
}
